// ParameterizedGAT_10153302687997
// MI455X (gfx1250) — matrix operand order repaired and checked by reading, not compiled, not run on hardware
//
#include <hip/hip_runtime.h>
#include <hip/hip_bf16.h>

#define N_NODES 50000      // NOTE: multiple of 16 (3125 row tiles) -> no M guards
#define N_EDGES 800000
#define F_IN    128
#define HEADS   4
#define C1      64
#define C2      32
#define NEG_SLOPE 0.2f
#define EPS_F   1e-16f

typedef __attribute__((ext_vector_type(16))) __bf16 v16bf;
typedef __attribute__((ext_vector_type(8)))  float  v8f;

// ---------------------------------------------------------------------------
// Ordered float atomic max via integer atomics (IEEE total order trick).
__device__ __forceinline__ void atomicMaxF32(float* addr, float v) {
    if (v >= 0.f) atomicMax((int*)addr, __float_as_int(v));
    else          atomicMin((unsigned int*)addr, __float_as_uint(v));
}

// ---------------------------------------------------------------------------
__global__ void fill_f32(float* __restrict__ p, long n, float v) {
    long i = (long)blockIdx.x * blockDim.x + threadIdx.x;
    if (i < n) p[i] = v;
}

// ---------------------------------------------------------------------------
// D[M x Ncol] = A[M x K] * B[K x Ncol]; f32 in memory, bf16 WMMA, f32 accum.
// Block = 256 threads (8 waves) -> 16 rows x 128 cols of D; one wave per
// 16x16 tile.  A (16xK) and B (Kx128, stored transposed) staged in LDS as
// bf16 with +2 element row padding (odd dword stride -> bank-conflict-free
// for both the staging stores and the per-lane ds_load_b128 reads).
// Requires: M % 16 == 0, Ncol % 128 == 0, K % 32 == 0.
template <int K>
__global__ __launch_bounds__(256) void gemm_bf16_wmma(
        const float* __restrict__ A, const float* __restrict__ B,
        float* __restrict__ D, int Ncol) {
    constexpr int KP = K + 2;                  // padded LDS row stride
    __shared__ unsigned short As[16 * KP];     // As[row][k]
    __shared__ unsigned short Bs[128 * KP];    // Bs[col][k]  (transposed)

    const int rowBase  = blockIdx.x * 16;
    const int colBlock = blockIdx.y * 128;
    const int tid = threadIdx.x;

    // Stage A slab: coalesced f32 reads, bf16 convert once per block.
    for (int i = tid; i < 16 * K; i += 256) {
        int r = i / K, c = i - r * K;
        As[r * KP + c] =
            __builtin_bit_cast(unsigned short, (__bf16)A[(long)(rowBase + r) * K + c]);
    }
    // Stage B slab transposed: read B row-major (coalesced), write Bs[col][k].
    for (int i = tid; i < K * 128; i += 256) {
        int k = i >> 7, c = i & 127;
        Bs[c * KP + k] =
            __builtin_bit_cast(unsigned short, (__bf16)B[(long)k * Ncol + colBlock + c]);
    }
    __syncthreads();

    const int wave = tid >> 5;
    const int lane = tid & 31;
    const int half = lane >> 4;              // lane group 0/1
    const int mr   = lane & 15;              // A row / B,D column within tile
    const int localCol = wave * 16 + mr;     // column within 128-col block

    const unsigned short* ap = &As[mr * KP];
    const unsigned short* bp = &Bs[localCol * KP];
    const int kbA = half * 8;                // ISA 16-bit A layout K-base
    const int kbB = half * 16;               // contiguous-order K-base; unused (B uses kbA)

    v8f acc = {};
#pragma unroll
    for (int k0 = 0; k0 < K; k0 += 32) {
        v16bf a, b;
#pragma unroll
        for (int i = 0; i < 16; ++i) {
            // A 16x32: elems 0-7 -> K = kbA+0..7, elems 8-15 -> +16  (2x b128)
            a[i] = __builtin_bit_cast(__bf16, ap[k0 + ((i >> 3) << 4) + kbA + (i & 7)]);
            // B 32x16: same K order as A: elems 0-7 -> K = kbA+0..7, elems 8-15 -> +16
            b[i] = __builtin_bit_cast(__bf16, bp[k0 + ((i >> 3) << 4) + kbA + (i & 7)]);
        }
        acc = __builtin_amdgcn_wmma_f32_16x16x32_bf16(
                  false, a, false, b, (short)0, acc, false, false);
    }

    // D f32 layout: VGPR r -> row r + 8*half, col = localCol.  M%16==0 -> no guard.
    float* dp = D + (long)(rowBase + half * 8) * Ncol + colBlock + localCol;
#pragma unroll
    for (int r = 0; r < 8; ++r) dp[(long)r * Ncol] = acc[r];
}

// ---------------------------------------------------------------------------
// a_s[n,h] = <h[n,h,:], att_src[h,:]> ; a_d likewise.  float4 vectorized.
__global__ void node_att(const float* __restrict__ h,
                         const float* __restrict__ att_s,
                         const float* __restrict__ att_d,
                         float* __restrict__ a_s, float* __restrict__ a_d,
                         int C) {
    long idx = (long)blockIdx.x * blockDim.x + threadIdx.x;
    if (idx >= (long)N_NODES * HEADS) return;
    int  hh   = (int)(idx & (HEADS - 1));
    long node = idx >> 2;                       // HEADS == 4
    const float* hp = h + (node * HEADS + hh) * C;
    const float* sp = att_s + hh * C;
    const float* dp = att_d + hh * C;
    float ss = 0.f, dd = 0.f;
    for (int c = 0; c < C; c += 4) {
        float4 hv = *(const float4*)(hp + c);
        float4 sv = *(const float4*)(sp + c);
        float4 dv = *(const float4*)(dp + c);
        ss += hv.x * sv.x + hv.y * sv.y + hv.z * sv.z + hv.w * sv.w;
        dd += hv.x * dv.x + hv.y * dv.y + hv.z * dv.z + hv.w * dv.w;
    }
    a_s[idx] = ss;
    a_d[idx] = dd;
}

// ---------------------------------------------------------------------------
// e = leaky_relu(a_s[src] + a_d[dst]); store logit; segment-max into m[dst].
__global__ void edge_logits_max(const int* __restrict__ src,
                                const int* __restrict__ dst,
                                const float* __restrict__ a_s,
                                const float* __restrict__ a_d,
                                float* __restrict__ elog,
                                float* __restrict__ m) {
    long idx = (long)blockIdx.x * blockDim.x + threadIdx.x;
    if (idx >= (long)N_EDGES * HEADS) return;
    long e  = idx >> 2;
    int  hh = (int)(idx & 3);
    int  s = src[e], d = dst[e];
    float v = a_s[(long)s * HEADS + hh] + a_d[(long)d * HEADS + hh];
    v = (v > 0.f) ? v : NEG_SLOPE * v;
    elog[idx] = v;
    atomicMaxF32(&m[(long)d * HEADS + hh], v);
}

// ---------------------------------------------------------------------------
// ex = exp(e - m[dst]); overwrite logit buffer; segment-sum into den[dst].
__global__ void edge_exp_sum(const int* __restrict__ dst,
                             float* __restrict__ elog,
                             const float* __restrict__ m,
                             float* __restrict__ den) {
    long idx = (long)blockIdx.x * blockDim.x + threadIdx.x;
    if (idx >= (long)N_EDGES * HEADS) return;
    long e  = idx >> 2;
    int  hh = (int)(idx & 3);
    int  d  = dst[e];
    float v = expf(elog[idx] - m[(long)d * HEADS + hh]);
    elog[idx] = v;
    atomicAdd(&den[(long)d * HEADS + hh], v);
}

// ---------------------------------------------------------------------------
// agg[dst,h,:] += (ex/(den[dst,h]+eps)) * h[src,h,:].  One thread per
// (edge, head, 4-channel chunk) -> float4 gathers, f32 L2 atomics.
__global__ void edge_aggregate(const int* __restrict__ src,
                               const int* __restrict__ dst,
                               const float* __restrict__ ex,
                               const float* __restrict__ den,
                               const float* __restrict__ h,
                               float* __restrict__ agg, int C) {
    const int  vec     = C >> 2;
    const long perEdge = (long)HEADS * vec;
    long idx = (long)blockIdx.x * blockDim.x + threadIdx.x;
    if (idx >= (long)N_EDGES * perEdge) return;
    long e  = idx / perEdge;
    int  r  = (int)(idx - e * perEdge);
    int  hh = r / vec;
    int  c4 = r - hh * vec;
    int  s = src[e], d = dst[e];
    float alpha = ex[e * HEADS + hh] / (den[(long)d * HEADS + hh] + EPS_F);
    const float4 hv = *(const float4*)(h + ((long)s * HEADS + hh) * C + c4 * 4);
    float* ap = agg + ((long)d * HEADS + hh) * C + c4 * 4;
    atomicAdd(ap + 0, alpha * hv.x);
    atomicAdd(ap + 1, alpha * hv.y);
    atomicAdd(ap + 2, alpha * hv.z);
    atomicAdd(ap + 3, alpha * hv.w);
}

// ---------------------------------------------------------------------------
// out[n,c] = act( mean_h agg[n,h,c] + bias[c] )
__global__ void finalize_mean_bias(const float* __restrict__ agg,
                                   const float* __restrict__ bias,
                                   float* __restrict__ out, int C, int do_relu) {
    long idx = (long)blockIdx.x * blockDim.x + threadIdx.x;
    if (idx >= (long)N_NODES * C) return;
    long node = idx / C;
    int  c    = (int)(idx - node * C);
    float ssum = 0.f;
#pragma unroll
    for (int hh = 0; hh < HEADS; ++hh)
        ssum += agg[(node * HEADS + hh) * C + c];
    float v = ssum * (1.f / HEADS) + bias[c];
    if (do_relu) v = fmaxf(v, 0.f);
    out[idx] = v;
}

// ---------------------------------------------------------------------------
static inline unsigned nblk(long n) { return (unsigned)((n + 255) / 256); }

extern "C" void kernel_launch(void* const* d_in, const int* in_sizes, int n_in,
                              void* d_out, int out_size, void* d_ws, size_t ws_size,
                              hipStream_t stream) {
    (void)in_sizes; (void)n_in; (void)out_size; (void)ws_size;

    const float* x    = (const float*)d_in[0];
    const int*   ei   = (const int*)  d_in[1];
    const float* W1   = (const float*)d_in[2];
    const float* as1  = (const float*)d_in[3];
    const float* ad1  = (const float*)d_in[4];
    const float* b1   = (const float*)d_in[5];
    const float* W2   = (const float*)d_in[6];
    const float* as2  = (const float*)d_in[7];
    const float* ad2  = (const float*)d_in[8];
    const float* b2   = (const float*)d_in[9];
    const int* src = ei;
    const int* dst = ei + N_EDGES;

    // Workspace layout (floats), reused across layers (~131 MB total).
    float* ws = (float*)d_ws;
    const long sz_h    = (long)N_NODES * HEADS * C1;   // 12.8M (covers layer2 too)
    const long sz_elog = (long)N_EDGES * HEADS;        //  3.2M
    float* hbuf  = ws;                                  // h1 / h2
    float* agg   = hbuf  + sz_h;                        // agg1 / agg2
    float* elog  = agg   + sz_h;                        // logits -> exp
    float* out1  = elog  + sz_elog;                     // relu'd layer-1 output
    float* a_s   = out1  + (long)N_NODES * C1;
    float* a_d   = a_s   + (long)N_NODES * HEADS;
    float* mbuf  = a_d   + (long)N_NODES * HEADS;
    float* den   = mbuf  + (long)N_NODES * HEADS;

    const long nh = (long)N_NODES * HEADS;
    const long eh = (long)N_EDGES * HEADS;

    // ---------------- Layer 1: K=128, Ncol=H*C1=256, C=64 -----------------
    fill_f32<<<nblk(nh), 256, 0, stream>>>(mbuf, nh, -__builtin_inff());
    fill_f32<<<nblk(nh), 256, 0, stream>>>(den,  nh, 0.f);
    fill_f32<<<nblk((long)N_NODES * HEADS * C1), 256, 0, stream>>>(
        agg, (long)N_NODES * HEADS * C1, 0.f);

    gemm_bf16_wmma<F_IN><<<dim3(N_NODES / 16, (HEADS * C1) / 128), 256, 0, stream>>>(
        x, W1, hbuf, HEADS * C1);

    node_att<<<nblk(nh), 256, 0, stream>>>(hbuf, as1, ad1, a_s, a_d, C1);
    edge_logits_max<<<nblk(eh), 256, 0, stream>>>(src, dst, a_s, a_d, elog, mbuf);
    edge_exp_sum<<<nblk(eh), 256, 0, stream>>>(dst, elog, mbuf, den);
    edge_aggregate<<<nblk((long)N_EDGES * HEADS * (C1 / 4)), 256, 0, stream>>>(
        src, dst, elog, den, hbuf, agg, C1);
    finalize_mean_bias<<<nblk((long)N_NODES * C1), 256, 0, stream>>>(
        agg, b1, out1, C1, 1);

    // ---------------- Layer 2: K=64, Ncol=H*C2=128, C=32 ------------------
    fill_f32<<<nblk(nh), 256, 0, stream>>>(mbuf, nh, -__builtin_inff());
    fill_f32<<<nblk(nh), 256, 0, stream>>>(den,  nh, 0.f);
    fill_f32<<<nblk((long)N_NODES * HEADS * C2), 256, 0, stream>>>(
        agg, (long)N_NODES * HEADS * C2, 0.f);

    gemm_bf16_wmma<C1><<<dim3(N_NODES / 16, (HEADS * C2) / 128), 256, 0, stream>>>(
        out1, W2, hbuf, HEADS * C2);

    node_att<<<nblk(nh), 256, 0, stream>>>(hbuf, as2, ad2, a_s, a_d, C2);
    edge_logits_max<<<nblk(eh), 256, 0, stream>>>(src, dst, a_s, a_d, elog, mbuf);
    edge_exp_sum<<<nblk(eh), 256, 0, stream>>>(dst, elog, mbuf, den);
    edge_aggregate<<<nblk((long)N_EDGES * HEADS * (C2 / 4)), 256, 0, stream>>>(
        src, dst, elog, den, hbuf, agg, C2);
    finalize_mean_bias<<<nblk((long)N_NODES * C2), 256, 0, stream>>>(
        agg, b2, (float*)d_out, C2, 0);
}
